// TFFunnelRelMultiheadAttention_55336358641847
// MI455X (gfx1250) — matrix operand order repaired and checked by reading, not compiled, not run on hardware
//
#include <hip/hip_runtime.h>

#define B_    2
#define S_    2048
#define D_    768
#define NH_   12
#define DH_   64
#define REL_  (2*S_)
#define INF_  1.0e6f
#define EPS_  1e-9f
#define SCALE_ 0.125f  // 1/sqrt(64)

typedef __bf16 bf16;
typedef __attribute__((ext_vector_type(8)))  __bf16 v8bf;
typedef __attribute__((ext_vector_type(16))) __bf16 v16bf;
typedef __attribute__((ext_vector_type(8)))  float  v8f;

union V16 { v16bf v; v8bf h[2]; };

__device__ inline bf16 f2bf(float f) {
  unsigned u = __builtin_bit_cast(unsigned, f);
  u += 0x7FFFu + ((u >> 16) & 1u);           // round-to-nearest-even
  unsigned short s = (unsigned short)(u >> 16);
  return __builtin_bit_cast(bf16, s);
}

__device__ inline v8f wmma_bf16(const V16& a, const V16& b, v8f c) {
  return __builtin_amdgcn_wmma_f32_16x16x32_bf16(false, a.v, false, b.v,
                                                 (short)0, c, false, false);
}

// ---------------------------------------------------------------- convert f32 -> bf16
__global__ void k_cvt(const float* __restrict__ in, bf16* __restrict__ out, int n) {
  int i = blockIdx.x * 256 + threadIdx.x;
  if (i < n) out[i] = f2bf(in[i]);
}

// ---------------------------------------------------------------- W[R][C] -> Wt[C][R] (bf16)
__global__ void k_transpose(const float* __restrict__ W, bf16* __restrict__ Wt, int R, int C) {
  int c = blockIdx.x * 16 + threadIdx.x;
  int r = blockIdx.y * 16 + threadIdx.y;
  if (r < R && c < C) Wt[(size_t)c * R + r] = f2bf(W[(size_t)r * C + c]);
}

// ---------------------------------------------------------------- generic 16x64-tile WMMA GEMM
// A: [M x 768] bf16 row-major.  Bt: [768 x 768] bf16, Bt[e][k] = W[k][e].
// One wave per 16(row) x 64(col) tile.  K = 768.
// MODE 0: q proj -> qw=(acc+r_w)*S, qr=(acc+r_r)*S (bf16) + tt_same/tt_diff dots with seg_embed
// MODE 1: k proj -> o0 = acc + bk (bf16)
// MODE 2: v proj -> oT[(b,hd,d)][i] = acc + bv (bf16, transposed over seq)
// MODE 3: r_head -> o0 = acc (bf16)
// MODE 4: post   -> f0 = acc + b_post + resid (f32)
template<int MODE>
__global__ __launch_bounds__(32)
void k_gemm(const bf16* __restrict__ A, const bf16* __restrict__ Bt,
            const float* __restrict__ p0, const float* __restrict__ p1,
            const float* __restrict__ p2, const float* __restrict__ p3,
            bf16* __restrict__ o0, bf16* __restrict__ o1, bf16* __restrict__ oT,
            float* __restrict__ f0, float* __restrict__ f1)
{
  const int K = D_;
  const int lane = threadIdx.x;
  const int col  = lane & 15;
  const int hf   = lane >> 4;
  const int rb   = hf * 8;                 // C-layout row offset for this half
  const size_t i0 = (size_t)blockIdx.x * 16;
  const int e0 = blockIdx.y * 64;

  v8f zero = {0,0,0,0,0,0,0,0};
  v8f acc[4] = {zero, zero, zero, zero};

  for (int k0 = 0; k0 < K; k0 += 32) {
    V16 a;
    const bf16* ap = A + (i0 + col) * K + k0 + hf * 16;
    a.h[0] = *(const v8bf*)(ap);
    a.h[1] = *(const v8bf*)(ap + 8);
#pragma unroll
    for (int nt = 0; nt < 4; ++nt) {
      V16 bb;
      const bf16* bp = Bt + (size_t)(e0 + nt * 16 + col) * K + k0 + hf * 16;
      bb.h[0] = *(const v8bf*)(bp);
      bb.h[1] = *(const v8bf*)(bp + 8);
      acc[nt] = wmma_bf16(a, bb, acc[nt]);
    }
  }

  if (MODE == 0) {
    const int hd = e0 >> 6;
    float ttS[8], ttD[8];
#pragma unroll
    for (int v = 0; v < 8; ++v) { ttS[v] = 0.f; ttD[v] = 0.f; }
#pragma unroll
    for (int nt = 0; nt < 4; ++nt) {
      int e  = e0 + nt * 16 + col;
      int eh = e & 63;
      float rw  = p0[hd * DH_ + eh];
      float rr  = p1[hd * DH_ + eh];
      float rs  = p2[hd * DH_ + eh];
      float sg0 = p3[0 * NH_ * DH_ + hd * DH_ + eh];  // diff
      float sg1 = p3[1 * NH_ * DH_ + hd * DH_ + eh];  // same
#pragma unroll
      for (int v = 0; v < 8; ++v) {
        float a = acc[nt][v];
        size_t g = i0 + v + rb;
        o0[g * (size_t)D_ + e] = f2bf((a + rw) * SCALE_);
        o1[g * (size_t)D_ + e] = f2bf((a + rr) * SCALE_);
        float qs = (a + rs) * SCALE_;
        ttD[v] += qs * sg0;
        ttS[v] += qs * sg1;
      }
    }
#pragma unroll
    for (int v = 0; v < 8; ++v) {
#pragma unroll
      for (int m = 1; m < 16; m <<= 1) {
        ttS[v] += __shfl_xor(ttS[v], m, 32);
        ttD[v] += __shfl_xor(ttD[v], m, 32);
      }
    }
    if (col == 0) {
#pragma unroll
      for (int v = 0; v < 8; ++v) {
        size_t g = i0 + v + rb;
        size_t b = g >> 11, i = g & (S_ - 1);
        f0[(b * NH_ + hd) * S_ + i] = ttS[v];
        f1[(b * NH_ + hd) * S_ + i] = ttD[v];
      }
    }
  } else if (MODE == 1 || MODE == 3) {
#pragma unroll
    for (int nt = 0; nt < 4; ++nt) {
      int e = e0 + nt * 16 + col;
      float bias = (MODE == 1) ? p0[e] : 0.f;
#pragma unroll
      for (int v = 0; v < 8; ++v) {
        size_t g = i0 + v + rb;
        o0[g * (size_t)D_ + e] = f2bf(acc[nt][v] + bias);
      }
    }
  } else if (MODE == 2) {
    const int hd = e0 >> 6;
#pragma unroll
    for (int nt = 0; nt < 4; ++nt) {
      int e = e0 + nt * 16 + col;
      int d = e & 63;
      float bias = p0[e];
#pragma unroll
      for (int v = 0; v < 8; ++v) {
        size_t g = i0 + v + rb;
        size_t b = g >> 11, i = g & (S_ - 1);
        oT[((b * NH_ + hd) * (size_t)DH_ + d) * S_ + i] = f2bf(acc[nt][v] + bias);
      }
    }
  } else {  // MODE 4: h = acc + b_post + query
#pragma unroll
    for (int nt = 0; nt < 4; ++nt) {
      int e = e0 + nt * 16 + col;
      float bias = p0[e];
#pragma unroll
      for (int v = 0; v < 8; ++v) {
        size_t g = i0 + v + rb;
        f0[g * (size_t)D_ + e] = acc[nt][v] + bias + p1[g * (size_t)D_ + e];
      }
    }
  }
}

// ---------------------------------------------------------------- fused flash attention
// One wave per (b, head, 16-row strip).  Streams j in chunks of 32 with online softmax.
// pos is produced from a 48-wide window of r_head via 3 WMMA tiles + LDS diagonal gather:
//   pos_shifted[i][j] = qr[i] . r_head[S + j - i]
__global__ __launch_bounds__(32)
void k_attn(const bf16* __restrict__ qw, const bf16* __restrict__ qr,
            const bf16* __restrict__ khd, const bf16* __restrict__ vtb,
            const bf16* __restrict__ rh,
            const float* __restrict__ ttSame, const float* __restrict__ ttDiff,
            const unsigned char* __restrict__ ttm, const float* __restrict__ amask,
            const float* __restrict__ clsm, bf16* __restrict__ avec)
{
  __shared__ __align__(16) float Wlds[16][48];
  __shared__ __align__(16) bf16  Plds[16][32];

  const int lane = threadIdx.x;
  const int col  = lane & 15;
  const int hf   = lane >> 4;
  const int rb   = hf * 8;

  int t = blockIdx.x;
  const int ib = t & (S_ / 16 - 1); t >>= 7;
  const int n  = t % NH_;
  const int b  = t / NH_;
  const int i0 = ib * 16;

  // A fragments for 16 q rows, K = 64 as two K=32 chunks
  V16 qwA[2], qrA[2];
#pragma unroll
  for (int kc = 0; kc < 2; ++kc) {
    const size_t base = ((size_t)(b * S_ + i0 + col)) * D_ + n * DH_ + kc * 32 + hf * 16;
    qwA[kc].h[0] = *(const v8bf*)(qw + base);
    qwA[kc].h[1] = *(const v8bf*)(qw + base + 8);
    qrA[kc].h[0] = *(const v8bf*)(qr + base);
    qrA[kc].h[1] = *(const v8bf*)(qr + base + 8);
  }

  float tts[8], ttd[8];
#pragma unroll
  for (int v = 0; v < 8; ++v) {
    int m = v + rb;
    tts[v] = ttSame[((size_t)b * NH_ + n) * S_ + i0 + m];
    ttd[v] = ttDiff[((size_t)b * NH_ + n) * S_ + i0 + m];
  }

  v8f zero = {0,0,0,0,0,0,0,0};
  float rmax[8], rsum[8];
  v8f O[4] = {zero, zero, zero, zero};
#pragma unroll
  for (int v = 0; v < 8; ++v) { rmax[v] = -1.0e30f; rsum[v] = 0.f; }

  for (int jc = 0; jc < S_; jc += 32) {
    // ---- content scores: two 16x16 tiles
    v8f c0 = zero, c1 = zero;
#pragma unroll
    for (int kc = 0; kc < 2; ++kc) {
      V16 bb;
      size_t kb = ((size_t)(b * S_ + jc + col)) * D_ + n * DH_ + kc * 32 + hf * 16;
      bb.h[0] = *(const v8bf*)(khd + kb);
      bb.h[1] = *(const v8bf*)(khd + kb + 8);
      c0 = wmma_bf16(qwA[kc], bb, c0);
      kb += (size_t)16 * D_;
      bb.h[0] = *(const v8bf*)(khd + kb);
      bb.h[1] = *(const v8bf*)(khd + kb + 8);
      c1 = wmma_bf16(qwA[kc], bb, c1);
    }

    // ---- pos window: W[m][ul] = qr[i0+m] . r_head[ubase+ul], ul in [0,48)
    const int ubase = S_ + jc - i0 - 16;
#pragma unroll
    for (int wt = 0; wt < 3; ++wt) {
      v8f w = zero;
#pragma unroll
      for (int kc = 0; kc < 2; ++kc) {
        V16 bb;
        size_t ub = ((size_t)(ubase + wt * 16 + col)) * D_ + n * DH_ + kc * 32 + hf * 16;
        bb.h[0] = *(const v8bf*)(rh + ub);
        bb.h[1] = *(const v8bf*)(rh + ub + 8);
        w = wmma_bf16(qrA[kc], bb, w);
      }
#pragma unroll
      for (int v = 0; v < 8; ++v) Wlds[v + rb][wt * 16 + col] = w[v];
    }
    __syncthreads();

    float pen0 = INF_ * (1.f - amask[(size_t)b * S_ + jc + col]);
    float pen1 = INF_ * (1.f - amask[(size_t)b * S_ + jc + 16 + col]);

    // ---- assemble scores: content + (pos + tt)*cls_mask - mask penalty
    float s0[8], s1[8];
#pragma unroll
    for (int v = 0; v < 8; ++v) {
      int m = v + rb;
      size_t i = i0 + m;
      {
        int jl = col; size_t j = jc + jl;
        float pos = Wlds[m][jl - m + 16];
        float cls = clsm[i * S_ + j];
        float tv  = ttm[((size_t)b * S_ + i) * S_ + j] ? tts[v] : ttd[v];
        s0[v] = c0[v] + (pos + tv) * cls - pen0;
      }
      {
        int jl = 16 + col; size_t j = jc + jl;
        float pos = Wlds[m][jl - m + 16];
        float cls = clsm[i * S_ + j];
        float tv  = ttm[((size_t)b * S_ + i) * S_ + j] ? tts[v] : ttd[v];
        s1[v] = c1[v] + (pos + tv) * cls - pen1;
      }
    }

    // ---- online softmax (row stats live in lanes of each 16-lane half)
    float p0[8], p1[8];
#pragma unroll
    for (int v = 0; v < 8; ++v) {
      float cm = fmaxf(s0[v], s1[v]);
#pragma unroll
      for (int msk = 1; msk < 16; msk <<= 1) cm = fmaxf(cm, __shfl_xor(cm, msk, 32));
      float nm = fmaxf(rmax[v], cm);
      float sc = __expf(rmax[v] - nm);
      p0[v] = __expf(s0[v] - nm);
      p1[v] = __expf(s1[v] - nm);
      float ls = p0[v] + p1[v];
#pragma unroll
      for (int msk = 1; msk < 16; msk <<= 1) ls += __shfl_xor(ls, msk, 32);
      rsum[v] = rsum[v] * sc + ls;
      rmax[v] = nm;
#pragma unroll
      for (int dt = 0; dt < 4; ++dt) O[dt][v] *= sc;
    }

    // ---- transpose P (C-layout -> A-layout) through LDS
    __syncthreads();
#pragma unroll
    for (int v = 0; v < 8; ++v) {
      Plds[v + rb][col]      = f2bf(p0[v]);
      Plds[v + rb][16 + col] = f2bf(p1[v]);
    }
    __syncthreads();
    V16 pa;
    {
      const bf16* pp = &Plds[col][hf * 16];
      pa.h[0] = *(const v8bf*)(pp);
      pa.h[1] = *(const v8bf*)(pp + 8);
    }

    // ---- P @ V : 4 d-tiles, B from transposed V (contiguous in j)
#pragma unroll
    for (int dt = 0; dt < 4; ++dt) {
      V16 bb;
      size_t vb = (((size_t)b * NH_ + n) * DH_ + dt * 16 + col) * S_ + jc + hf * 16;
      bb.h[0] = *(const v8bf*)(vtb + vb);
      bb.h[1] = *(const v8bf*)(vtb + vb + 8);
      O[dt] = wmma_bf16(pa, bb, O[dt]);
    }
    __syncthreads();
  }

  float inv[8];
#pragma unroll
  for (int v = 0; v < 8; ++v) inv[v] = 1.f / rsum[v];
#pragma unroll
  for (int dt = 0; dt < 4; ++dt) {
#pragma unroll
    for (int v = 0; v < 8; ++v) {
      size_t g = (size_t)b * S_ + i0 + v + rb;
      avec[g * (size_t)D_ + n * DH_ + dt * 16 + col] = f2bf(O[dt][v] * inv[v]);
    }
  }
}

// ---------------------------------------------------------------- LayerNorm over D=768
__global__ void k_ln(const float* __restrict__ h, const float* __restrict__ gamma,
                     const float* __restrict__ beta, float* __restrict__ out)
{
  __shared__ float red[16];
  __shared__ float mu_sh, rs_sh;
  const int row = blockIdx.x;
  const int tid = threadIdx.x;  // 256 threads
  const float* hr = h + (size_t)row * D_;
  float x0 = hr[tid], x1 = hr[tid + 256], x2 = hr[tid + 512];
  float s = x0 + x1 + x2;
  float q = x0 * x0 + x1 * x1 + x2 * x2;
#pragma unroll
  for (int m = 1; m < 32; m <<= 1) { s += __shfl_xor(s, m, 32); q += __shfl_xor(q, m, 32); }
  if ((tid & 31) == 0) { red[tid >> 5] = s; red[8 + (tid >> 5)] = q; }
  __syncthreads();
  if (tid == 0) {
    float ts = 0.f, tq = 0.f;
    for (int i = 0; i < 8; ++i) { ts += red[i]; tq += red[8 + i]; }
    float mu = ts / (float)D_;
    mu_sh = mu;
    rs_sh = rsqrtf(tq / (float)D_ - mu * mu + EPS_);
  }
  __syncthreads();
  float mu = mu_sh, rstd = rs_sh;
  float* orow = out + (size_t)row * D_;
  orow[tid]       = (x0 - mu) * rstd * gamma[tid]       + beta[tid];
  orow[tid + 256] = (x1 - mu) * rstd * gamma[tid + 256] + beta[tid + 256];
  orow[tid + 512] = (x2 - mu) * rstd * gamma[tid + 512] + beta[tid + 512];
}

// ---------------------------------------------------------------- host
extern "C" void kernel_launch(void* const* d_in, const int* in_sizes, int n_in,
                              void* d_out, int out_size, void* d_ws, size_t ws_size,
                              hipStream_t stream)
{
  const float* query = (const float*)d_in[0];
  const float* key   = (const float*)d_in[1];
  const float* value = (const float*)d_in[2];
  const float* r     = (const float*)d_in[3];
  const unsigned char* ttm = (const unsigned char*)d_in[4];  // bool
  const float* amask = (const float*)d_in[5];
  const float* clsm  = (const float*)d_in[6];
  const float* wq    = (const float*)d_in[7];
  const float* wk    = (const float*)d_in[8];
  const float* bk    = (const float*)d_in[9];
  const float* wv    = (const float*)d_in[10];
  const float* bv    = (const float*)d_in[11];
  const float* r_w   = (const float*)d_in[12];
  const float* r_r   = (const float*)d_in[13];
  const float* r_s   = (const float*)d_in[14];
  const float* rker  = (const float*)d_in[15];
  const float* seg   = (const float*)d_in[16];
  const float* wpost = (const float*)d_in[17];
  const float* bpost = (const float*)d_in[18];
  const float* gamma = (const float*)d_in[19];
  const float* beta  = (const float*)d_in[20];
  float* out = (float*)d_out;

  char* ws = (char*)d_ws;
  size_t off = 0;
  auto alloc = [&](size_t bytes) -> char* {
    off = (off + 255) & ~(size_t)255;
    char* p = ws + off;
    off += bytes;
    return p;
  };

  const size_t MD = (size_t)B_ * S_ * D_;      // 3,145,728
  const size_t RD = (size_t)REL_ * D_;         // 3,145,728
  const size_t WD = (size_t)D_ * D_;

  bf16* q_bf  = (bf16*)alloc(MD * 2);
  bf16* k_bf  = (bf16*)alloc(MD * 2);
  bf16* v_bf  = (bf16*)alloc(MD * 2);
  bf16* r_bf  = (bf16*)alloc(RD * 2);
  bf16* wqT   = (bf16*)alloc(WD * 2);
  bf16* wkT   = (bf16*)alloc(WD * 2);
  bf16* wvT   = (bf16*)alloc(WD * 2);
  bf16* rkT   = (bf16*)alloc(WD * 2);
  bf16* wpT   = (bf16*)alloc(WD * 2);
  bf16* qwb   = (bf16*)alloc(MD * 2);
  bf16* qrb   = (bf16*)alloc(MD * 2);
  bf16* khead = (bf16*)alloc(MD * 2);
  bf16* vtb   = (bf16*)alloc(MD * 2);
  bf16* rhead = (bf16*)alloc(RD * 2);
  bf16* avec  = (bf16*)alloc(MD * 2);
  float* ttS  = (float*)alloc((size_t)B_ * NH_ * S_ * 4);
  float* ttD  = (float*)alloc((size_t)B_ * NH_ * S_ * 4);
  // f32 h buffer aliases dead bf16 input copies (q_bf/k_bf/v_bf: 18MB >= 12.6MB)
  float* hbuf = (float*)q_bf;

  // 1) convert inputs to bf16
  k_cvt<<<dim3((int)((MD + 255) / 256)), 256, 0, stream>>>(query, q_bf, (int)MD);
  k_cvt<<<dim3((int)((MD + 255) / 256)), 256, 0, stream>>>(key,   k_bf, (int)MD);
  k_cvt<<<dim3((int)((MD + 255) / 256)), 256, 0, stream>>>(value, v_bf, (int)MD);
  k_cvt<<<dim3((int)((RD + 255) / 256)), 256, 0, stream>>>(r,     r_bf, (int)RD);

  // 2) transpose + convert weights: Wt[e][k] = W[k][e]
  dim3 tb(16, 16), tg(D_ / 16, D_ / 16);
  k_transpose<<<tg, tb, 0, stream>>>(wq,    wqT, D_, D_);
  k_transpose<<<tg, tb, 0, stream>>>(wk,    wkT, D_, D_);
  k_transpose<<<tg, tb, 0, stream>>>(wv,    wvT, D_, D_);
  k_transpose<<<tg, tb, 0, stream>>>(rker,  rkT, D_, D_);
  k_transpose<<<tg, tb, 0, stream>>>(wpost, wpT, D_, D_);

  // 3) projection GEMMs (WMMA bf16)
  dim3 gg(B_ * S_ / 16, D_ / 64);                         // (256, 12)
  k_gemm<0><<<gg, 32, 0, stream>>>(q_bf, wqT, r_w, r_r, r_s, seg,
                                   qwb, qrb, nullptr, ttS, ttD);
  k_gemm<1><<<gg, 32, 0, stream>>>(k_bf, wkT, bk, nullptr, nullptr, nullptr,
                                   khead, nullptr, nullptr, nullptr, nullptr);
  k_gemm<2><<<gg, 32, 0, stream>>>(v_bf, wvT, bv, nullptr, nullptr, nullptr,
                                   nullptr, nullptr, vtb, nullptr, nullptr);
  dim3 gr(REL_ / 16, D_ / 64);
  k_gemm<3><<<gr, 32, 0, stream>>>(r_bf, rkT, nullptr, nullptr, nullptr, nullptr,
                                   rhead, nullptr, nullptr, nullptr, nullptr);

  // 4) fused attention (flash-style, rel-shift folded into r_head indexing)
  k_attn<<<B_ * NH_ * (S_ / 16), 32, 0, stream>>>(qwb, qrb, khead, vtb, rhead,
                                                  ttS, ttD, ttm, amask, clsm, avec);

  // 5) post GEMM + bias + residual
  k_gemm<4><<<gg, 32, 0, stream>>>(avec, wpT, bpost, query, nullptr, nullptr,
                                   nullptr, nullptr, nullptr, hbuf, nullptr);

  // 6) LayerNorm
  k_ln<<<B_ * S_, 256, 0, stream>>>(hbuf, gamma, beta, out);
}
